// FP8PureLinear_31147102831044
// MI455X (gfx1250) — 8-bit matrix operand order repaired and checked by reading, compiled, not run on hardware
//
#include <hip/hip_runtime.h>


typedef __attribute__((ext_vector_type(16))) int   v16i;
typedef __attribute__((ext_vector_type(8)))  float v8f;

union AFrag { v16i v; unsigned long long d[8]; };
union BFrag { v16i v; uint4 q[4]; };

__device__ __forceinline__ float bf16_bits_to_f32(unsigned int h) {
    return __uint_as_float(h << 16);
}

__device__ __forceinline__ unsigned short f32_to_bf16_rne(float f) {
    unsigned int u = __float_as_uint(f);
    unsigned int r = u + 0x7FFFu + ((u >> 16) & 1u);
    return (unsigned short)(r >> 16);
}

__device__ __forceinline__ void async_copy16(unsigned lds_off, const unsigned char* g) {
    asm volatile("global_load_async_to_lds_b128 %0, %1, off"
                 :: "v"(lds_off), "v"((unsigned long long)(uintptr_t)g)
                 : "memory");
}
__device__ __forceinline__ void wait_async0() {
    asm volatile("s_wait_asynccnt 0" ::: "memory");
}

__device__ __forceinline__ unsigned char f32_to_e4m3_sw(float x) {
    unsigned int u = __float_as_uint(x);
    unsigned int s = (u >> 24) & 0x80u;
    float a = __builtin_fabsf(x);
    if (!(a > 0.0f))   return (unsigned char)s;
    if (a >= 448.0f)   return (unsigned char)(s | 0x7E);
    int e;
    (void)__builtin_frexpf(a, &e);
    int ue = e - 1;
    if (ue < -6) {
        int q = (int)__builtin_rintf(__builtin_ldexpf(a, 9));
        if (q >= 8) return (unsigned char)(s | 0x08);
        return (unsigned char)(s | q);
    }
    int q = (int)__builtin_rintf(__builtin_ldexpf(a, 3 - ue));
    if (q == 16) { q = 8; ++ue; }
    if (ue > 8) return (unsigned char)(s | 0x7E);
    return (unsigned char)(s | ((ue + 7) << 3) | (q & 7));
}

__device__ __forceinline__ unsigned int pack4_e4m3(float f0, float f1, float f2, float f3) {
#if __has_builtin(__builtin_amdgcn_cvt_pk_fp8_f32)
    int r = 0;
    r = __builtin_amdgcn_cvt_pk_fp8_f32(f0, f1, r, false);
    r = __builtin_amdgcn_cvt_pk_fp8_f32(f2, f3, r, true);
    return (unsigned int)r;
#else
    return (unsigned int)f32_to_e4m3_sw(f0)
         | ((unsigned int)f32_to_e4m3_sw(f1) << 8)
         | ((unsigned int)f32_to_e4m3_sw(f2) << 16)
         | ((unsigned int)f32_to_e4m3_sw(f3) << 24);
#endif
}

__global__ void init_ws_kernel(unsigned int* amax) {
    if (threadIdx.x < 2) amax[threadIdx.x] = 0u;
}

__global__ void amax_abs_bf16_kernel(const unsigned int* __restrict__ p, int n_u32,
                                     unsigned int* __restrict__ dst) {
    float m = 0.0f;
    int stride = gridDim.x * blockDim.x;
    for (int i = blockIdx.x * blockDim.x + threadIdx.x; i < n_u32; i += stride) {
        unsigned int v = p[i];
        float a = __builtin_fabsf(bf16_bits_to_f32(v & 0xFFFFu));
        float b = __builtin_fabsf(bf16_bits_to_f32(v >> 16));
        m = fmaxf(m, fmaxf(a, b));
    }
    for (int off = 16; off > 0; off >>= 1)
        m = fmaxf(m, __shfl_xor(m, off, 32));
    __shared__ float red[8];
    int lane = threadIdx.x & 31, w = threadIdx.x >> 5;
    if (lane == 0) red[w] = m;
    __syncthreads();
    if (threadIdx.x < 8) {
        m = red[threadIdx.x];
        for (int off = 4; off > 0; off >>= 1)
            m = fmaxf(m, __shfl_xor(m, off, 32));
        if (threadIdx.x == 0)
            atomicMax(dst, __float_as_uint(m));
    }
}

__global__ void quant_e4m3_kernel(const uint2* __restrict__ in, unsigned int* __restrict__ out,
                                  int n4, const unsigned int* __restrict__ amax_bits) {
    float amax  = __uint_as_float(*amax_bits);
    float scale = fmaxf(amax * (1.0f / 448.0f), 1e-12f);
    float inv   = 1.0f / scale;
    int stride = gridDim.x * blockDim.x;
    for (int i = blockIdx.x * blockDim.x + threadIdx.x; i < n4; i += stride) {
        uint2 v = in[i];
        float f0 = bf16_bits_to_f32(v.x & 0xFFFFu) * inv;
        float f1 = bf16_bits_to_f32(v.x >> 16)     * inv;
        float f2 = bf16_bits_to_f32(v.y & 0xFFFFu) * inv;
        float f3 = bf16_bits_to_f32(v.y >> 16)     * inv;
        out[i] = pack4_e4m3(f0, f1, f2, f3);
    }
}

#define LDT 144

__global__ __launch_bounds__(256)
void gemm_fp8_wmma_kernel(const unsigned char* __restrict__ A,
                          const unsigned char* __restrict__ B,
                          unsigned short* __restrict__ out,
                          const unsigned int* __restrict__ amax,
                          int K, int N) {
    __shared__ __align__(16) unsigned char lA[2][128 * LDT];
    __shared__ __align__(16) unsigned char lB[2][128 * LDT];

    const int t    = threadIdx.x;
    const int lane = t & 31;
    const int wave = t >> 5;
    const int wm   = wave >> 2;
    const int wn   = wave & 3;
    const int lr   = lane & 15;
    const int hi   = lane >> 4;

    const unsigned char* Ap = A + (size_t)(blockIdx.y * 128) * (size_t)K;
    const unsigned char* Bp = B + (size_t)(blockIdx.x * 128) * (size_t)K;

    const int srow = t >> 3;
    const int skc  = (t & 7) << 4;

    const unsigned ldsA0 = (unsigned)(uintptr_t)&lA[0][0];
    const unsigned ldsA1 = (unsigned)(uintptr_t)&lA[1][0];
    const unsigned ldsB0 = (unsigned)(uintptr_t)&lB[0][0];
    const unsigned ldsB1 = (unsigned)(uintptr_t)&lB[1][0];

    v8f acc[4][2];
    #pragma unroll
    for (int i = 0; i < 4; ++i)
        #pragma unroll
        for (int j = 0; j < 2; ++j)
            acc[i][j] = (v8f){0.f, 0.f, 0.f, 0.f, 0.f, 0.f, 0.f, 0.f};

    auto stage = [&](int k0, unsigned la, unsigned lb) {
        #pragma unroll
        for (int i = 0; i < 4; ++i) {
            int row = srow + (i << 5);
            async_copy16(la + row * LDT + skc, Ap + (size_t)row * K + k0 + skc);
            async_copy16(lb + row * LDT + skc, Bp + (size_t)row * K + k0 + skc);
        }
    };

    const int nk = K >> 7;
    stage(0, ldsA0, ldsB0);
    wait_async0();
    __syncthreads();

    for (int kt = 0; kt < nk; ++kt) {
        const int cur = kt & 1;
        const unsigned char* bufA = lA[cur];
        const unsigned char* bufB = lB[cur];

        if (kt + 1 < nk)
            stage((kt + 1) << 7, cur ? ldsA0 : ldsA1, cur ? ldsB0 : ldsB1);

        AFrag a[4];
        #pragma unroll
        for (int tm = 0; tm < 4; ++tm) {
            const unsigned char* base = &bufA[(wm * 64 + tm * 16 + lr) * LDT + hi * 8];
            #pragma unroll
            for (int c = 0; c < 8; ++c)
                a[tm].d[c] = *(const unsigned long long*)(base + c * 16);
        }
        BFrag b[2];
        #pragma unroll
        for (int tn = 0; tn < 2; ++tn) {
            const unsigned char* base = &bufB[(wn * 32 + tn * 16 + lr) * LDT + hi * 16];
            #pragma unroll
            for (int c = 0; c < 4; ++c) {
                ((unsigned long long*)&b[tn].q[c])[0] = *(const unsigned long long*)(base - hi * 8 + c * 32);
                ((unsigned long long*)&b[tn].q[c])[1] = *(const unsigned long long*)(base - hi * 8 + c * 32 + 16);
            }
        }

        #pragma unroll
        for (int tm = 0; tm < 4; ++tm)
            #pragma unroll
            for (int tn = 0; tn < 2; ++tn)
                acc[tm][tn] = __builtin_amdgcn_wmma_f32_16x16x128_fp8_fp8(
                    a[tm].v, b[tn].v, (short)0, acc[tm][tn],
                     false,  false);

        wait_async0();
        __syncthreads();
    }

    float ax = __uint_as_float(amax[0]);
    float aw = __uint_as_float(amax[1]);
    float s  = fmaxf(ax * (1.0f / 448.0f), 1e-12f) * fmaxf(aw * (1.0f / 448.0f), 1e-12f);

    const int rowBase = blockIdx.y * 128 + wm * 64;
    const int colBase = blockIdx.x * 128 + wn * 32;
    #pragma unroll
    for (int tm = 0; tm < 4; ++tm) {
        #pragma unroll
        for (int tn = 0; tn < 2; ++tn) {
            int cc = colBase + tn * 16 + lr;
            #pragma unroll
            for (int r = 0; r < 8; ++r) {
                int rr = rowBase + tm * 16 + hi * 8 + r;
                out[(size_t)rr * N + cc] = f32_to_bf16_rne(acc[tm][tn][r] * s);
            }
        }
    }
}

extern "C" void kernel_launch(void* const* d_in, const int* in_sizes, int n_in,
                              void* d_out, int out_size, void* d_ws, size_t ws_size,
                              hipStream_t stream) {
    const unsigned short* x = (const unsigned short*)d_in[0];
    const unsigned short* w = (const unsigned short*)d_in[1];
    unsigned short* out = (unsigned short*)d_out;

    const int xN = in_sizes[0];
    const int wN = in_sizes[1];
    int K = 1;
    while ((long long)K * K < (long long)wN) K <<= 1;
    const int N = wN / K;
    const int M = xN / K;

    unsigned char* ws   = (unsigned char*)d_ws;
    unsigned int*  amax = (unsigned int*)ws;
    unsigned char* xq   = ws + 256;
    unsigned char* wq   = xq + (size_t)M * (size_t)K;

    init_ws_kernel<<<1, 32, 0, stream>>>(amax);

    amax_abs_bf16_kernel<<<1024, 256, 0, stream>>>((const unsigned int*)x, xN / 2, amax + 0);
    amax_abs_bf16_kernel<<<512,  256, 0, stream>>>((const unsigned int*)w, wN / 2, amax + 1);

    quant_e4m3_kernel<<<2048, 256, 0, stream>>>((const uint2*)x, (unsigned int*)xq, xN / 4, amax + 0);
    quant_e4m3_kernel<<<512,  256, 0, stream>>>((const uint2*)w, (unsigned int*)wq, wN / 4, amax + 1);

    dim3 grid(N / 128, M / 128);
    gemm_fp8_wmma_kernel<<<grid, 256, 0, stream>>>(xq, wq, out, amax, K, N);
}
